// Attention_79353815761006
// MI455X (gfx1250) — matrix operand order repaired and checked by reading, not compiled, not run on hardware
//
#include <hip/hip_runtime.h>

// ---------------------------------------------------------------------------
// Shapes (fixed by the reference): B=2, S=2048, HID=2048, H=32, KV=8, D=64
// ---------------------------------------------------------------------------
#define BB   2
#define SS   2048
#define HID  2048
#define NH   32
#define NKV  8
#define DD   64

typedef __attribute__((ext_vector_type(16))) __bf16 v16bf;
typedef __attribute__((ext_vector_type(8)))  __bf16 v8bf;
typedef __attribute__((ext_vector_type(8)))  float  v8f;
typedef __attribute__((ext_vector_type(4)))  unsigned int u32x4;
typedef __attribute__((ext_vector_type(4)))  int i32x4;
typedef __attribute__((ext_vector_type(8)))  int i32x8;

#if defined(__has_builtin)
#  if __has_builtin(__builtin_amdgcn_tensor_load_to_lds)
#    define USE_TDM 1
#  else
#    define USE_TDM 0
#  endif
#else
#  define USE_TDM 0
#endif

__device__ __forceinline__ unsigned short f2bf(float f) {
  unsigned int u = __float_as_uint(f);
  u += 0x7fffu + ((u >> 16) & 1u);          // round-to-nearest-even
  return (unsigned short)(u >> 16);
}

__device__ __forceinline__ v16bf cat16(v8bf lo, v8bf hi) {
  return __builtin_shufflevector(lo, hi, 0,1,2,3,4,5,6,7,8,9,10,11,12,13,14,15);
}

// A-operand (16x32 bf16, MxK): lane holds row M=lane%16.
// Lanes 0-15 hold K = {0..7, 16..23}, lanes 16-31 K = {8..15, 24..31}.
__device__ __forceinline__ v16bf loadA(const unsigned short* row, int grp) {
  v8bf lo = *(const v8bf*)(row + grp * 8);
  v8bf hi = *(const v8bf*)(row + 16 + grp * 8);
  return cat16(lo, hi);
}

// B-operand (32x16 bf16, KxN): lane holds column N=lane%16.
// Lanes 0-15 hold K = {0..7, 16..23}, lanes 16-31 K = {8..15, 24..31}.
__device__ __forceinline__ v16bf loadB(const unsigned short* col, int grp) {
  v8bf lo = *(const v8bf*)(col + grp * 8);
  v8bf hi = *(const v8bf*)(col + 16 + grp * 8);
  return cat16(lo, hi);
}

__device__ __forceinline__ v8f wmma_bf16(v16bf a, v16bf b, v8f c) {
  return __builtin_amdgcn_wmma_f32_16x16x32_bf16(false, a, false, b,
                                                 (short)0, c, false, false);
}

// ---------------------------------------------------------------------------
// Kernel 1: f32 -> bf16 elementwise
// ---------------------------------------------------------------------------
__global__ void cvt_bf16_kernel(const float* __restrict__ src,
                                unsigned short* __restrict__ dst, int n) {
  int i = blockIdx.x * blockDim.x + threadIdx.x;
  if (i < n) dst[i] = f2bf(src[i]);
}

// ---------------------------------------------------------------------------
// Kernel 2: f32 [K][N] -> bf16 [N][K], 32x32 LDS-tiled so both the loads and
// the stores are coalesced. K, N are multiples of 32.
// ---------------------------------------------------------------------------
__global__ void cvt_transpose_kernel(const float* __restrict__ src,
                                     unsigned short* __restrict__ dst,
                                     int K, int N) {
  __shared__ unsigned short t[32][33];
  const int tn = blockIdx.x % (N / 32);
  const int tk = blockIdx.x / (N / 32);
  const int x  = threadIdx.x;               // 0..31
  const int y0 = threadIdx.y;               // 0..7
#pragma unroll
  for (int r = 0; r < 4; ++r) {
    int y = y0 + r * 8;
    t[y][x] = f2bf(src[(size_t)(tk * 32 + y) * N + tn * 32 + x]);
  }
  __syncthreads();
#pragma unroll
  for (int r = 0; r < 4; ++r) {
    int y = y0 + r * 8;
    dst[(size_t)(tn * 32 + y) * K + tk * 32 + x] = t[x][y];
  }
}

// ---------------------------------------------------------------------------
// Kernel 3: fused QKV projection + RoPE.
// One wave computes a 16(s) x 64(d) tile = one head-slot of one row tile.
// slot 0..31 -> Q head, 32..39 -> K head, 40..47 -> V head.
// Q scaled by 1/sqrt(D)=0.125 at store time. V stored transposed [d][s]
// through an LDS tile so the global stores are 16B contiguous.
// ---------------------------------------------------------------------------
__global__ void qkv_rope_kernel(const unsigned short* __restrict__ xb,
                                const unsigned short* __restrict__ wqt,
                                const unsigned short* __restrict__ wkt,
                                const unsigned short* __restrict__ wvt,
                                const float* __restrict__ cosT,
                                const float* __restrict__ sinT,
                                unsigned short* __restrict__ qws,
                                unsigned short* __restrict__ kws,
                                unsigned short* __restrict__ vtws) {
  __shared__ __align__(16) unsigned short vstage[4][64 * 16];  // 8 KB

  const int lane   = threadIdx.x & 31;
  const int widx   = threadIdx.x >> 5;
  const int lanelo = lane & 15;
  const int grp    = lane >> 4;

  const int tile  = blockIdx.x * 4 + widx;          // 0 .. 12287
  const int slot  = tile % 48;
  const int stile = (tile / 48) % (SS / 16);
  const int b     = tile / (48 * (SS / 16));

  const unsigned short* wt;
  int head, kind;                                    // 0=Q 1=K 2=V
  if (slot < 32)      { kind = 0; head = slot;      wt = wqt; }
  else if (slot < 40) { kind = 1; head = slot - 32; wt = wkt; }
  else                { kind = 2; head = slot - 40; wt = wvt; }

  const unsigned short* arow =
      xb + (size_t)(b * SS + stile * 16 + lanelo) * HID;
  const unsigned short* brow =
      wt + (size_t)(head * DD + lanelo) * HID;

  v8f zero = {};
  v8f acc[4] = {zero, zero, zero, zero};

  for (int kk = 0; kk < HID; kk += 32) {
    __builtin_prefetch(arow + kk + 128, 0, 1);       // global_prefetch
    v16bf a = loadA(arow + kk, grp);
#pragma unroll
    for (int j = 0; j < 4; ++j) {
      const unsigned short* bc = brow + (size_t)(j * 16) * HID + kk;
      v16bf bm = loadB(bc, grp);
      acc[j] = wmma_bf16(a, bm, acc[j]);
    }
  }

  const int srow = stile * 16 + 8 * grp;             // + v gives global s

  if (kind != 2) {                                   // RoPE for Q and K
    float nc[4][8];
#pragma unroll
    for (int j = 0; j < 4; ++j) {
      const int d = j * 16 + lanelo;
#pragma unroll
      for (int v = 0; v < 8; ++v) {
        const int s = srow + v;
        float c  = cosT[(size_t)s * DD + d];
        float sn = sinT[(size_t)s * DD + d];
        float cur  = acc[j][v];
        float pair = (j < 2) ? -acc[j + 2][v] : acc[j - 2][v];
        nc[j][v] = cur * c + pair * sn;
      }
    }
#pragma unroll
    for (int j = 0; j < 4; ++j)
#pragma unroll
      for (int v = 0; v < 8; ++v) acc[j][v] = nc[j][v];
  }

  if (kind == 0) {
    unsigned short* qb = qws + (size_t)(b * NH + head) * SS * DD;
#pragma unroll
    for (int j = 0; j < 4; ++j)
#pragma unroll
      for (int v = 0; v < 8; ++v)
        qb[(size_t)(srow + v) * DD + j * 16 + lanelo] =
            f2bf(acc[j][v] * 0.125f);
  } else if (kind == 1) {
    unsigned short* kb = kws + (size_t)(b * NKV + head) * SS * DD;
#pragma unroll
    for (int j = 0; j < 4; ++j)
#pragma unroll
      for (int v = 0; v < 8; ++v)
        kb[(size_t)(srow + v) * DD + j * 16 + lanelo] = f2bf(acc[j][v]);
  } else {
    // V: transpose 16(s) x 64(d) tile through LDS, then contiguous stores.
    unsigned short* vs = vstage[widx];
#pragma unroll
    for (int j = 0; j < 4; ++j)
#pragma unroll
      for (int v = 0; v < 8; ++v)
        vs[(j * 16 + lanelo) * 16 + (v + 8 * grp)] = f2bf(acc[j][v]);
    asm volatile("s_wait_dscnt 0x0" ::: "memory");   // intra-wave DS ordering
    unsigned short* vb = vtws + (size_t)(b * NKV + head) * DD * SS;
#pragma unroll
    for (int r = 0; r < 2; ++r) {
      const int d = lane * 2 + r;                    // 0..63
      const unsigned short* sr = vs + d * 16;
      unsigned short* dr = vb + (size_t)d * SS + stile * 16;
      *(v8bf*)(dr)     = *(const v8bf*)(sr);
      *(v8bf*)(dr + 8) = *(const v8bf*)(sr + 8);
    }
  }
}

// ---------------------------------------------------------------------------
// Kernel 4: causal flash attention. One wave owns a 16-row Q tile of one
// (b, h). K-loop in chunks of 32 keys: scores = 4 WMMA, online softmax with
// wave32 xor-shuffle row reductions, P staged through LDS (C-layout store ->
// s_wait_dscnt -> A-layout ds_load_b128), P*V = 4 WMMA into f32 accumulators.
// ---------------------------------------------------------------------------
__global__ void flash_attn_kernel(const unsigned short* __restrict__ qws,
                                  const unsigned short* __restrict__ kws,
                                  const unsigned short* __restrict__ vtws,
                                  unsigned short* __restrict__ ows) {
  __shared__ __align__(16) unsigned short pbuf[4][16 * 32];

  const int lane   = threadIdx.x & 31;
  const int widx   = threadIdx.x >> 5;
  const int lanelo = lane & 15;
  const int grp    = lane >> 4;

  const int tile = blockIdx.x * 4 + widx;            // 0 .. 8191
  const int qt   = tile % (SS / 16);
  const int h    = (tile / (SS / 16)) % NH;
  const int b    = tile / ((SS / 16) * NH);
  const int kvh  = h >> 2;                           // repeat_interleave(4)

  const unsigned short* qrow =
      qws + (size_t)((b * NH + h) * SS + qt * 16 + lanelo) * DD;
  const unsigned short* kmat = kws + (size_t)(b * NKV + kvh) * SS * DD;
  const unsigned short* vtm  = vtws + (size_t)(b * NKV + kvh) * DD * SS;

  v16bf qa0 = loadA(qrow, grp);
  v16bf qa1 = loadA(qrow + 32, grp);

  float m[8], l[8];
  v8f zero = {};
  v8f o[4] = {zero, zero, zero, zero};
#pragma unroll
  for (int v = 0; v < 8; ++v) { m[v] = -3.0e38f; l[v] = 0.0f; }

  unsigned short* pb = pbuf[widx];
  const int nchunks = (qt * 16) / 32 + 1;

  for (int c = 0; c < nchunks; ++c) {
    const int kb0 = c * 32;

    // ---- scores: (16q x 32k) = Q(16x64) * K^T(64x32) ----
    v8f sc[2];
#pragma unroll
    for (int j = 0; j < 2; ++j) {
      const unsigned short* kr =
          kmat + (size_t)(kb0 + j * 16 + lanelo) * DD;
      v16bf b0 = loadB(kr, grp);
      v16bf b1 = loadB(kr + 32, grp);
      v8f s = zero;
      s = wmma_bf16(qa0, b0, s);
      s = wmma_bf16(qa1, b1, s);
      sc[j] = s;
    }

    // ---- causal mask ----
#pragma unroll
    for (int j = 0; j < 2; ++j) {
      const int col = kb0 + j * 16 + lanelo;
#pragma unroll
      for (int v = 0; v < 8; ++v) {
        const int row = qt * 16 + 8 * grp + v;
        if (col > row) sc[j][v] = -3.0e38f;
      }
    }

    // ---- online softmax (rows live across 16-lane halves) ----
#pragma unroll
    for (int v = 0; v < 8; ++v) {
      float cmax = fmaxf(sc[0][v], sc[1][v]);
      cmax = fmaxf(cmax, __shfl_xor(cmax, 1, 32));
      cmax = fmaxf(cmax, __shfl_xor(cmax, 2, 32));
      cmax = fmaxf(cmax, __shfl_xor(cmax, 4, 32));
      cmax = fmaxf(cmax, __shfl_xor(cmax, 8, 32));
      float mnew  = fmaxf(m[v], cmax);
      float scale = __expf(m[v] - mnew);
      float p0 = __expf(sc[0][v] - mnew);
      float p1 = __expf(sc[1][v] - mnew);
      float rs = p0 + p1;
      rs += __shfl_xor(rs, 1, 32);
      rs += __shfl_xor(rs, 2, 32);
      rs += __shfl_xor(rs, 4, 32);
      rs += __shfl_xor(rs, 8, 32);
      l[v] = l[v] * scale + rs;
      m[v] = mnew;
#pragma unroll
      for (int t = 0; t < 4; ++t) o[t][v] *= scale;
      const int r = 8 * grp + v;                     // C-layout row
      pb[r * 32 + lanelo]      = f2bf(p0);
      pb[r * 32 + 16 + lanelo] = f2bf(p1);
    }

    asm volatile("s_wait_dscnt 0x0" ::: "memory");

    // ---- P*V: O(16x64) += P(16x32) * V(32x64) ----
    const unsigned short* pr = pb + lanelo * 32;
    v16bf pa = loadA(pr, grp);
#pragma unroll
    for (int t = 0; t < 4; ++t) {
      const unsigned short* vr =
          vtm + (size_t)(t * 16 + lanelo) * SS + kb0;
      v16bf vb = loadB(vr, grp);
      o[t] = wmma_bf16(pa, vb, o[t]);
    }
  }

  unsigned short* ob = ows + (size_t)b * SS * (NH * DD);
#pragma unroll
  for (int v = 0; v < 8; ++v) {
    const float inv = 1.0f / l[v];
    const int s = qt * 16 + 8 * grp + v;
#pragma unroll
    for (int t = 0; t < 4; ++t)
      ob[(size_t)s * (NH * DD) + h * DD + t * 16 + lanelo] =
          f2bf(o[t][v] * inv);
  }
}

// ---------------------------------------------------------------------------
// Kernel 5: output projection (attn_out @ wo) -> f32 d_out.
// Block = 64(s) x 64(n) tile, 4 waves of 16 rows each. The shared 64x32 bf16
// weight tile is staged into double-buffered LDS: via the Tensor Data Mover
// (tensor_load_to_lds + s_wait_tensorcnt) when the builtin exists, else a
// cooperative global->LDS copy. WMMA B operands then come from ds_load_b128.
// ---------------------------------------------------------------------------
__global__ void out_proj_kernel(const unsigned short* __restrict__ ao,
                                const unsigned short* __restrict__ wot,
                                float* __restrict__ out) {
  __shared__ __align__(16) unsigned short bstage[2][64 * 32];  // 8 KB

  const int lane   = threadIdx.x & 31;
  const int widx   = threadIdx.x >> 5;
  const int lanelo = lane & 15;
  const int grp    = lane >> 4;

  const int ntile = blockIdx.x % (HID / 64);
  const int s64   = (blockIdx.x / (HID / 64)) % (SS / 64);
  const int b     = blockIdx.x / ((HID / 64) * (SS / 64));
  const int nbase = ntile * 64;

  const unsigned short* arow =
      ao + (size_t)(b * SS + s64 * 64 + widx * 16 + lanelo) * HID;

  v8f zero = {};
  v8f acc[4] = {zero, zero, zero, zero};

  auto stage = [&](int kk, int par) {
#if USE_TDM
    if (widx == 0) {
      unsigned long long g =
          (unsigned long long)(wot + (size_t)nbase * HID + kk);
      unsigned ldsoff = (unsigned)(unsigned long long)(&bstage[par][0]);
      // D# group0: count=1 | lds_addr | global_addr[56:0] | type=2
      u32x4 g0 = { 1u, ldsoff,
                   (unsigned)(g & 0xffffffffu),
                   ((unsigned)((g >> 32) & 0x1ffffffu)) | (2u << 30) };
      // D# group1: data_size=2B; tensor_dim0/1 = 1<<20 (never OOB);
      // tile_dim0=32 (k), tile_dim1=64 (n rows); tensor_dim0_stride=HID.
      i32x8 g1 = { (int)0x00010000,   // [17:16] data_size=1 (2 bytes)
                   0,                 // atomic barrier addr
                   (int)0x00000010,   // tensor_dim0[31:16]=0x10 (=1<<20)
                   (int)0x00200010,   // tensor_dim1 hi | tile_dim0=32
                   64,                // tile_dim1=64
                   (int)HID,          // tensor_dim0_stride lo
                   0, 0 };
      i32x4 g2 = { 0, 0, 0, 0 };
      i32x4 g3 = { 0, 0, 0, 0 };
#  if __clang_major__ >= 23
      i32x8 g4 = { 0, 0, 0, 0, 0, 0, 0, 0 };
      __builtin_amdgcn_tensor_load_to_lds(g0, g1, g2, g3, g4, 0);
#  else
      __builtin_amdgcn_tensor_load_to_lds(g0, g1, g2, g3, 0);
#  endif
    }
#else
    // cooperative copy: 256 x 16B chunks across 128 threads
    const int t = threadIdx.x;
#pragma unroll
    for (int r = 0; r < 2; ++r) {
      int c  = t + r * 128;
      int n  = c >> 2;
      int ki = (c & 3) * 8;
      *(v8bf*)&bstage[par][n * 32 + ki] =
          *(const v8bf*)(wot + (size_t)(nbase + n) * HID + kk + ki);
    }
#endif
  };

  stage(0, 0);
  for (int i = 0; i < HID / 32; ++i) {
    const int kk = i * 32;
#if USE_TDM
    if (widx == 0) __builtin_amdgcn_s_wait_tensorcnt(0);  // chunk i landed
#endif
    if (i + 1 < HID / 32) stage(kk + 32, (i + 1) & 1);    // overlap next DMA
    __syncthreads();                                      // buf[i&1] ready
    const unsigned short* bbuf = &bstage[i & 1][0];
    v16bf a = loadA(arow + kk, grp);
#pragma unroll
    for (int j = 0; j < 4; ++j) {
      v16bf bm = loadB(bbuf + (j * 16 + lanelo) * 32, grp);
      acc[j] = wmma_bf16(a, bm, acc[j]);
    }
    __syncthreads();                                      // reads done
  }

  float* ob = out + (size_t)b * SS * HID;
#pragma unroll
  for (int v = 0; v < 8; ++v) {
    const int s = s64 * 64 + widx * 16 + 8 * grp + v;
#pragma unroll
    for (int j = 0; j < 4; ++j)
      ob[(size_t)s * HID + nbase + j * 16 + lanelo] = acc[j][v];
  }
}

// ---------------------------------------------------------------------------
// Launch
// ---------------------------------------------------------------------------
extern "C" void kernel_launch(void* const* d_in, const int* in_sizes, int n_in,
                              void* d_out, int out_size, void* d_ws,
                              size_t ws_size, hipStream_t stream) {
  (void)in_sizes; (void)n_in; (void)out_size; (void)ws_size;
  const float* x    = (const float*)d_in[0];
  const float* cosT = (const float*)d_in[1];
  const float* sinT = (const float*)d_in[2];
  const float* wq   = (const float*)d_in[3];
  const float* wk   = (const float*)d_in[4];
  const float* wv   = (const float*)d_in[5];
  const float* wo   = (const float*)d_in[6];
  float* out = (float*)d_out;

  char* ws = (char*)d_ws;
  size_t off = 0;
  auto take = [&](size_t bytes) {
    char* p = ws + off;
    off += (bytes + 255) & ~(size_t)255;
    return p;
  };
  unsigned short* xb   = (unsigned short*)take((size_t)BB * SS * HID * 2);
  unsigned short* wqt  = (unsigned short*)take((size_t)HID * (NH * DD) * 2);
  unsigned short* wkt  = (unsigned short*)take((size_t)HID * (NKV * DD) * 2);
  unsigned short* wvt  = (unsigned short*)take((size_t)HID * (NKV * DD) * 2);
  unsigned short* wot  = (unsigned short*)take((size_t)(NH * DD) * HID * 2);
  unsigned short* qws  = (unsigned short*)take((size_t)BB * NH * SS * DD * 2);
  unsigned short* kws  = (unsigned short*)take((size_t)BB * NKV * SS * DD * 2);
  unsigned short* vtws = (unsigned short*)take((size_t)BB * NKV * DD * SS * 2);
  unsigned short* aws  = (unsigned short*)take((size_t)BB * SS * (NH * DD) * 2);

  const int nx = BB * SS * HID;
  cvt_bf16_kernel<<<(nx + 255) / 256, 256, 0, stream>>>(x, xb, nx);

  dim3 tb(32, 8);
  cvt_transpose_kernel<<<(HID / 32) * ((NH * DD) / 32), tb, 0, stream>>>(
      wq, wqt, HID, NH * DD);
  cvt_transpose_kernel<<<(HID / 32) * ((NKV * DD) / 32), tb, 0, stream>>>(
      wk, wkt, HID, NKV * DD);
  cvt_transpose_kernel<<<(HID / 32) * ((NKV * DD) / 32), tb, 0, stream>>>(
      wv, wvt, HID, NKV * DD);
  cvt_transpose_kernel<<<((NH * DD) / 32) * (HID / 32), tb, 0, stream>>>(
      wo, wot, NH * DD, HID);

  qkv_rope_kernel<<<(BB * (SS / 16) * 48) / 4, 128, 0, stream>>>(
      xb, wqt, wkt, wvt, cosT, sinT, qws, kws, vtws);
  flash_attn_kernel<<<(BB * NH * (SS / 16)) / 4, 128, 0, stream>>>(
      qws, kws, vtws, aws);
  out_proj_kernel<<<BB * (SS / 64) * (HID / 64), 128, 0, stream>>>(
      aws, wot, out);
}
